// CrossAttention_19267223290450
// MI455X (gfx1250) — compile-verified, matrix operand order checked by reading, not run on hardware
//
#include <hip/hip_runtime.h>
#include <hip/hip_bf16.h>

// ---------------------------------------------------------------------------
// CDNA5 (gfx1250, wave32) cross-attention:
//   q = x@Wq ; k = ctx@Wk ; v = ctx@Wv ; attn ; out = attn@Wo + bo
// All matmuls use v_wmma_f32_16x16x32_bf16 (f32 accumulate).
// ---------------------------------------------------------------------------

typedef __attribute__((ext_vector_type(16))) __bf16 v16bf;
typedef __attribute__((ext_vector_type(8)))  float  v8f;
typedef __attribute__((ext_vector_type(4)))  int    v4i;

union Frag {
    v16bf v;
    unsigned int u[8];
};

__device__ __forceinline__ v8f wmma_bf16(v16bf a, v16bf b, v8f c) {
    // (neg_a, A, neg_b, B, c_mod, C, reuse_a, reuse_b)
    return __builtin_amdgcn_wmma_f32_16x16x32_bf16(false, a, false, b,
                                                   (short)0, c, false, false);
}

__device__ __forceinline__ unsigned short f2bf(float f) {
    unsigned int u = __float_as_uint(f);
    u += 0x7FFFu + ((u >> 16) & 1u);          // round-to-nearest-even
    return (unsigned short)(u >> 16);
}

#if __has_builtin(__builtin_amdgcn_cvt_pk_bf16_f32)
__device__ __forceinline__ unsigned int pack2bf(float lo, float hi) {
    auto r = __builtin_amdgcn_cvt_pk_bf16_f32(lo, hi);   // v_cvt_pk_bf16_f32
    unsigned int u;
    __builtin_memcpy(&u, &r, 4);
    return u;
}
#else
__device__ __forceinline__ unsigned int pack2bf(float lo, float hi) {
    return (unsigned int)f2bf(lo) | ((unsigned int)f2bf(hi) << 16);
}
#endif

// Async global->LDS DMA path (CDNA5), guarded so both toolchains compile.
#if __has_builtin(__builtin_amdgcn_global_load_async_to_lds_b128)
#define HAVE_ASYNC_LDS 1
typedef __attribute__((address_space(1))) v4i* gv4p;   // global int4*
typedef __attribute__((address_space(3))) v4i* lv4p;   // LDS int4*
#else
#define HAVE_ASYNC_LDS 0
#endif

// ---------------------------------------------------------------------------
// Generic tiled WMMA GEMM: C[M,N] = A[M,K] * B[K,N]
// Block = 256 threads = 8 waves; tile 128x128, BK=32.
// Wave grid 4(M) x 2(N); each wave owns 32x64 = 2x4 WMMA tiles.
// A is f32 (converted) or bf16; B is always f32 (weights), converted on stage.
// EPI: 0 = bf16 row-major, 1 = f32 + bias, 2 = K layout [b,h,j(80),64] bf16,
//      3 = V layout [b,h,64,j(96)] bf16  (EPI 2/3: M rows map to (b = r/77, j = r%77))
// ---------------------------------------------------------------------------
#define BM 128
#define BN 128
#define BK 32
#define LDS_STRIDE 40   // halfs per row: 32 data + 8 pad (keeps 16B alignment, skews banks)

template<bool ABF16, int EPI>
__global__ __launch_bounds__(256) void gemm_kernel(
    const void* __restrict__ Ap, const float* __restrict__ Bp,
    void* __restrict__ Outp, const float* __restrict__ bias,
    int M, int N, int K)
{
    __shared__ unsigned short Ash[BM * LDS_STRIDE];
    __shared__ unsigned short Bsh[BN * LDS_STRIDE];

    const int tid  = threadIdx.x;
    const int lane = tid & 31;
    const int wave = tid >> 5;
    const int ln   = lane & 15;
    const int half = lane >> 4;
    const int kb   = half * 8;        // per-half K offset of WMMA fragments
    const int wm   = wave & 3;        // 0..3 along M
    const int wn   = wave >> 2;       // 0..1 along N
    const long long row0 = (long long)blockIdx.y * BM;
    const int       col0 = blockIdx.x * BN;

    v8f zero = {};
    v8f acc[2][4];
    for (int ms = 0; ms < 2; ms++)
        for (int ns = 0; ns < 4; ns++) acc[ms][ns] = zero;

    for (int kt = 0; kt < K; kt += BK) {
        // ---- stage A tile (128 x 32) into LDS as bf16 ----
        if constexpr (!ABF16) {
            const float* A = (const float*)Ap;
            for (int i = 0; i < 4; i++) {
                int f  = tid + 256 * i;           // 1024 float4 per tile
                int r  = f >> 3;
                int c4 = f & 7;
                long long gr = row0 + r;
                float4 val = make_float4(0.f, 0.f, 0.f, 0.f);
                if (gr < M) {
                    const float* src = A + gr * (long long)K + kt + c4 * 4;
                    val = *(const float4*)src;
                    if (kt + BK < K) __builtin_prefetch(src + BK, 0, 1);
                }
                uint2 pk;
                pk.x = pack2bf(val.x, val.y);
                pk.y = pack2bf(val.z, val.w);
                *(uint2*)&Ash[r * LDS_STRIDE + c4 * 4] = pk;
            }
        } else {
            const unsigned short* A = (const unsigned short*)Ap;
            for (int i = 0; i < 2; i++) {
                int f  = tid + 256 * i;           // 512 groups of 8 halfs
                int r  = f >> 2;
                int c8 = (f & 3) * 8;
                long long gr = row0 + r;
                const unsigned short* src = A + gr * (long long)K + kt + c8;
#if HAVE_ASYNC_LDS
                // CDNA5 async DMA: global -> LDS without VGPR round trip (ASYNCcnt)
                __builtin_amdgcn_global_load_async_to_lds_b128(
                    (gv4p)(void*)src,
                    (lv4p)(void*)&Ash[r * LDS_STRIDE + c8],
                    0, 0);
#else
                uint4 val = *(const uint4*)src;
                *(uint4*)&Ash[r * LDS_STRIDE + c8] = val;
#endif
                if (kt + BK < K) __builtin_prefetch(src + BK, 0, 1);
            }
        }
        // ---- stage B tile (32 x 128), transposed into LDS: Bsh[n][k] ----
        for (int i = 0; i < 4; i++) {
            int f  = tid + 256 * i;
            int kr = f >> 5;
            int c4 = f & 31;
            const float* src = Bp + (long long)(kt + kr) * N + col0 + c4 * 4;
            float4 val = *(const float4*)src;
            if (kt + BK < K) __builtin_prefetch(src + (long long)BK * N, 0, 1);
            int nl = c4 * 4;
            Bsh[(nl + 0) * LDS_STRIDE + kr] = f2bf(val.x);
            Bsh[(nl + 1) * LDS_STRIDE + kr] = f2bf(val.y);
            Bsh[(nl + 2) * LDS_STRIDE + kr] = f2bf(val.z);
            Bsh[(nl + 3) * LDS_STRIDE + kr] = f2bf(val.w);
        }
#if HAVE_ASYNC_LDS
        if constexpr (ABF16) {
            asm volatile("s_wait_asynccnt 0" ::: "memory");
        }
#endif
        __syncthreads();

        // ---- fragment loads (per ISA 16-bit A layout: lane half picks k base) ----
        Frag af[2], bf[4];
        for (int ms = 0; ms < 2; ms++) {
            const unsigned short* ar = &Ash[(wm * 32 + ms * 16 + ln) * LDS_STRIDE];
            for (int v = 0; v < 4; v++) {
                af[ms].u[v]     = *(const unsigned int*)(ar + kb + 2 * v);
                af[ms].u[4 + v] = *(const unsigned int*)(ar + kb + 16 + 2 * v);
            }
        }
        for (int ns = 0; ns < 4; ns++) {
            const unsigned short* br = &Bsh[(wn * 64 + ns * 16 + ln) * LDS_STRIDE];
            for (int v = 0; v < 4; v++) {
                bf[ns].u[v]     = *(const unsigned int*)(br + kb + 2 * v);
                bf[ns].u[4 + v] = *(const unsigned int*)(br + kb + 16 + 2 * v);
            }
        }
        for (int ms = 0; ms < 2; ms++)
            for (int ns = 0; ns < 4; ns++)
                acc[ms][ns] = wmma_bf16(af[ms].v, bf[ns].v, acc[ms][ns]);
        __syncthreads();
    }

    // ---- epilogue (C layout: lane -> n = ln, vgpr v -> m = v + 8*half) ----
    for (int ms = 0; ms < 2; ms++)
        for (int ns = 0; ns < 4; ns++)
            for (int v = 0; v < 8; v++) {
                long long r = row0 + wm * 32 + ms * 16 + v + 8 * half;
                int       c = col0 + wn * 64 + ns * 16 + ln;
                float   val = acc[ms][ns][v];
                if constexpr (EPI == 0) {
                    ((unsigned short*)Outp)[r * (long long)N + c] = f2bf(val);
                } else if constexpr (EPI == 1) {
                    ((float*)Outp)[r * (long long)N + c] = val + bias[c];
                } else if constexpr (EPI == 2) {
                    if (r < M) {
                        int b = (int)(r / 77), j = (int)(r % 77);
                        int h = c >> 6, d = c & 63;
                        ((unsigned short*)Outp)[(((long long)(b * 16 + h)) * 80 + j) * 64 + d] = f2bf(val);
                    }
                } else {
                    if (r < M) {
                        int b = (int)(r / 77), j = (int)(r % 77);
                        int h = c >> 6, d = c & 63;
                        ((unsigned short*)Outp)[(((long long)(b * 16 + h)) * 64 + d) * 96 + j] = f2bf(val);
                    }
                }
            }
}

// ---------------------------------------------------------------------------
// Attention: one wave handles (b, h, 16 query rows).
//   S = Q(16x64) * K^T  -> 5 n-tiles (Nc=77 padded to 80), 2 WMMA k-steps
//   masked softmax (butterfly reductions inside each 16-lane half)
//   P staged to LDS bf16 (16 x 96, zero padded)
//   O = P(16x96) * V(96x64) -> 3 k-steps x 4 n-tiles
// Output written bf16 in-place over the wave's own Q slice.
// ---------------------------------------------------------------------------
#define NCPAD 96

__global__ __launch_bounds__(256) void attn_kernel(
    const unsigned short* __restrict__ Qbuf,
    const unsigned short* __restrict__ Kbuf,
    const unsigned short* __restrict__ Vbuf,
    unsigned short* __restrict__ Abuf)
{
    __shared__ unsigned short Psh[8 * 16 * NCPAD];

    const int tid  = threadIdx.x;
    const int lane = tid & 31;
    const int wave = tid >> 5;
    const int ln   = lane & 15;
    const int half = lane >> 4;
    const int kb   = half * 8;
    const int pbase = wave * (16 * NCPAD);

    long long gw = (long long)blockIdx.x * 8 + wave;  // 0 .. 65535
    int mt = (int)(gw % 256);
    int bh = (int)(gw / 256);
    int h  = bh % 16;
    int b  = bh / 16;
    int n0 = mt * 16;

    // zero the LDS pad columns [80..95] for all 16 rows (one b128 store/lane)
    {
        int mrow = ln;
        int c    = 80 + half * 8;
        *(uint4*)&Psh[pbase + mrow * NCPAD + c] = make_uint4(0u, 0u, 0u, 0u);
    }

    // ---- load Q fragments straight from global (row slice = one 128B line) ----
    Frag aq[2];
    {
        const unsigned short* qr =
            Qbuf + ((long long)(b * 4096 + n0 + ln)) * 1024 + h * 64;
        for (int v = 0; v < 4; v++) {
            aq[0].u[v]     = *(const unsigned int*)(qr + kb + 2 * v);
            aq[0].u[4 + v] = *(const unsigned int*)(qr + kb + 16 + 2 * v);
            aq[1].u[v]     = *(const unsigned int*)(qr + 32 + kb + 2 * v);
            aq[1].u[4 + v] = *(const unsigned int*)(qr + 32 + kb + 16 + 2 * v);
        }
    }

    // ---- S = Q * K^T ----
    v8f zero = {};
    v8f accS[5];
    const unsigned short* Kb = Kbuf + ((long long)(b * 16 + h)) * 80 * 64;
    for (int t = 0; t < 5; t++) {
        accS[t] = zero;
        const unsigned short* kr = Kb + (t * 16 + ln) * 64;
        Frag bk0, bk1;
        for (int v = 0; v < 4; v++) {
            bk0.u[v]     = *(const unsigned int*)(kr + kb + 2 * v);
            bk0.u[4 + v] = *(const unsigned int*)(kr + kb + 16 + 2 * v);
            bk1.u[v]     = *(const unsigned int*)(kr + 32 + kb + 2 * v);
            bk1.u[4 + v] = *(const unsigned int*)(kr + 32 + kb + 16 + 2 * v);
        }
        accS[t] = wmma_bf16(aq[0].v, bk0.v, accS[t]);
        accS[t] = wmma_bf16(aq[1].v, bk1.v, accS[t]);
    }

    // ---- masked softmax; row m = v + 8*half lives in one 16-lane half ----
    const float SCALE_ = 0.125f;   // 64^-0.5
    float rinv[8];
    for (int v = 0; v < 8; v++) {
        float sv[5];
        float mx = -1e30f;
        for (int t = 0; t < 5; t++) {
            int n = t * 16 + ln;
            sv[t] = accS[t][v] * SCALE_;
            if (n < 77) mx = fmaxf(mx, sv[t]);
        }
        for (int off = 1; off < 16; off <<= 1)
            mx = fmaxf(mx, __shfl_xor(mx, off));
        float sm = 0.f;
        int mrow = v + 8 * half;
        for (int t = 0; t < 5; t++) {
            int n = t * 16 + ln;
            float p = (n < 77) ? __expf(sv[t] - mx) : 0.f;
            sm += p;
            Psh[pbase + mrow * NCPAD + n] = f2bf(p);
        }
        for (int off = 1; off < 16; off <<= 1)
            sm += __shfl_xor(sm, off);
        rinv[v] = 1.0f / sm;
    }

    // same-wave LDS ops are in-order; belt-and-braces wait before re-reading P
    asm volatile("s_wait_dscnt 0" ::: "memory");

    // ---- O = P * V ----
    v8f accO[4];
    for (int ns = 0; ns < 4; ns++) accO[ns] = zero;
    const unsigned short* Vb = Vbuf + ((long long)(b * 16 + h)) * 64 * 96;
    for (int ks = 0; ks < 3; ks++) {
        Frag ap;
        const unsigned short* pr = &Psh[pbase + ln * NCPAD + ks * 32];
        for (int v = 0; v < 4; v++) {
            ap.u[v]     = *(const unsigned int*)(pr + kb + 2 * v);
            ap.u[4 + v] = *(const unsigned int*)(pr + kb + 16 + 2 * v);
        }
        for (int ns = 0; ns < 4; ns++) {
            Frag bv;
            const unsigned short* vr = Vb + (ns * 16 + ln) * 96 + ks * 32;
            for (int v = 0; v < 4; v++) {
                bv.u[v]     = *(const unsigned int*)(vr + kb + 2 * v);
                bv.u[4 + v] = *(const unsigned int*)(vr + kb + 16 + 2 * v);
            }
            accO[ns] = wmma_bf16(ap.v, bv.v, accO[ns]);
        }
    }

    // ---- normalize by row sum and store bf16 (in-place over own Q slice) ----
    for (int ns = 0; ns < 4; ns++)
        for (int v = 0; v < 8; v++) {
            int mrow = v + 8 * half;
            int d    = ns * 16 + ln;
            float o  = accO[ns][v] * rinv[v];
            Abuf[((long long)(b * 4096 + n0 + mrow)) * 1024 + h * 64 + d] = f2bf(o);
        }
}

// ---------------------------------------------------------------------------
// Host-side orchestration (graph-capture safe: only kernel launches + memsetAsync)
// ---------------------------------------------------------------------------
extern "C" void kernel_launch(void* const* d_in, const int* in_sizes, int n_in,
                              void* d_out, int out_size, void* d_ws, size_t ws_size,
                              hipStream_t stream) {
    const float* x   = (const float*)d_in[0];   // [16,4096,1024]
    const float* ctx = (const float*)d_in[1];   // [16,77,768]
    const float* Wq  = (const float*)d_in[2];   // [1024,1024]
    const float* Wk  = (const float*)d_in[3];   // [768,1024]
    const float* Wv  = (const float*)d_in[4];   // [768,1024]
    const float* Wo  = (const float*)d_in[5];   // [1024,1024]
    const float* bo  = (const float*)d_in[6];   // [1024]
    float* out = (float*)d_out;

    char* ws = (char*)d_ws;
    const size_t qBytes = (size_t)16 * 4096 * 1024 * 2;      // 134 MB bf16 Q / attn-out (in-place)
    const size_t kBytes = (size_t)16 * 16 * 80 * 64 * 2;     // K: [b,h,80,64]
    const size_t vBytes = (size_t)16 * 16 * 64 * 96 * 2;     // V^T: [b,h,64,96]
    unsigned short* Qbuf = (unsigned short*)(ws);
    unsigned short* Kbuf = (unsigned short*)(ws + qBytes);
    unsigned short* Vbuf = (unsigned short*)(ws + qBytes + kBytes);

    // zero padded K/V staging (ws is poisoned by the harness)
    (void)hipMemsetAsync(ws + qBytes, 0, kBytes + vBytes, stream);

    dim3 blk(256);
    // K / V projections: M=1232 (16*77), K=768, N=1024
    dim3 gKV(1024 / BN, (1232 + BM - 1) / BM);
    gemm_kernel<false, 2><<<gKV, blk, 0, stream>>>(ctx, Wk, Kbuf, nullptr, 1232, 1024, 768);
    gemm_kernel<false, 3><<<gKV, blk, 0, stream>>>(ctx, Wv, Vbuf, nullptr, 1232, 1024, 768);

    // Q projection: M=65536, K=1024, N=1024 -> bf16 row-major
    dim3 gQ(1024 / BN, 65536 / BM);
    gemm_kernel<false, 0><<<gQ, blk, 0, stream>>>(x, Wq, Qbuf, nullptr, 65536, 1024, 1024);

    // attention: 65536 (b,h,16-row) wave tasks, 8 waves per block
    attn_kernel<<<65536 / 8, blk, 0, stream>>>(Qbuf, Kbuf, Vbuf, Qbuf);

    // output projection + bias: A is bf16 (attn out), M=65536, K=1024, N=1024
    gemm_kernel<true, 1><<<gQ, blk, 0, stream>>>(Qbuf, Wo, out, bo, 65536, 1024, 1024);
}
